// GNNBlock_15178414424582
// MI455X (gfx1250) — matrix operand order repaired and checked by reading, not compiled, not run on hardware
//
#include <hip/hip_runtime.h>
#include <hip/hip_bf16.h>

typedef __attribute__((ext_vector_type(8)))  __bf16 v8bf;
typedef __attribute__((ext_vector_type(16))) __bf16 v16bf;
typedef __attribute__((ext_vector_type(8)))  float  v8f;

#define NN      100000
#define NE      1000000
#define IN_CH   312
#define IN_PAD  320
#define HID     256

// ---------------- utility kernels ----------------

__global__ void zero_f32(float* __restrict__ p, int n) {
  int i = blockIdx.x * blockDim.x + threadIdx.x;
  if (i < n) p[i] = 0.0f;
}

// one wave per edge: gather x[src] and atomically accumulate into agg[dst]
__global__ void scatter_edges(const float* __restrict__ x,
                              const long long* __restrict__ src,
                              const long long* __restrict__ dst,
                              float* __restrict__ agg,
                              float* __restrict__ deg,
                              int E, int C, int addDeg) {
  int gw   = (blockIdx.x * blockDim.x + threadIdx.x) >> 5;
  int lane = threadIdx.x & 31;
  if (gw >= E) return;
  int s = (int)src[gw];
  int d = (int)dst[gw];
  const float* xs = x + (long)s * C;
  float*       ad = agg + (long)d * C;
  if (addDeg && lane == 0) atomicAdd(&deg[d], 1.0f);
  for (int c = lane; c < C; c += 32) atomicAdd(&ad[c], xs[c]);
}

__global__ void make_invdeg(float* __restrict__ deg, int n) {
  int i = blockIdx.x * blockDim.x + threadIdx.x;
  if (i < n) deg[i] = 1.0f / fmaxf(deg[i], 1.0f);
}

// meanb[i][c] = bf16(agg[i][c] * inv[i]), zero-padded to Cpad
__global__ void mean_to_bf16(const float* __restrict__ agg,
                             const float* __restrict__ inv,
                             __bf16* __restrict__ out,
                             int C, int Cpad, int n) {
  int i = blockIdx.x * blockDim.x + threadIdx.x;
  int total = n * Cpad;
  if (i >= total) return;
  int row = i / Cpad, c = i - row * Cpad;
  float v = (c < C) ? agg[(long)row * C + c] * inv[row] : 0.0f;
  out[i] = (__bf16)v;
}

// out[i][c] = bf16(in[i][c]), zero-padded to Cpad
__global__ void f32_to_bf16_pad(const float* __restrict__ in,
                                __bf16* __restrict__ out,
                                int C, int Cpad, int n) {
  int i = blockIdx.x * blockDim.x + threadIdx.x;
  int total = n * Cpad;
  if (i >= total) return;
  int row = i / Cpad, c = i - row * Cpad;
  out[i] = (__bf16)((c < C) ? in[(long)row * C + c] : 0.0f);
}

// WT[n][k] = bf16(W[k][n]); K zero-padded to Kpad. W is [K x Ncols] row-major.
__global__ void weight_T_bf16(const float* __restrict__ W,
                              __bf16* __restrict__ WT,
                              int K, int Kpad, int Ncols) {
  int i = blockIdx.x * blockDim.x + threadIdx.x;
  int total = Ncols * Kpad;
  if (i >= total) return;
  int nn = i / Kpad, k = i - nn * Kpad;
  WT[i] = (__bf16)((k < K) ? W[(long)k * Ncols + nn] : 0.0f);
}

// ---------------- fused SAGE GEMM ----------------
// C[m][n] = act( A1[m][:] @ B1 + A2[m][:] @ B2 + bias[n] )
// A*: [M x K*p] bf16 row-major (K*p multiple of 32, zero padded)
// B*T: [256 x K*p] bf16 row-major (transposed weights)
// Writes f32 C (stride 256) and optionally bf16 copy Cb.
// Each wave: one 16x16 tile. 8 waves/block -> 8 N-tiles at same M-tile.
__global__ __launch_bounds__(256)
void sage_gemm(const __bf16* __restrict__ A1, const __bf16* __restrict__ B1T, int K1p,
               const __bf16* __restrict__ A2, const __bf16* __restrict__ B2T, int K2p,
               const float* __restrict__ bias,
               float* __restrict__ Cout, __bf16* __restrict__ Cb,
               int doRelu) {
  const int lane = threadIdx.x & 31;
  const int wid  = threadIdx.x >> 5;
  const int m0   = blockIdx.x * 16;
  const int n0   = (blockIdx.y * 8 + wid) * 16;
  const int r    = lane & 15;   // A: row within tile; B/C: column within tile
  const int h    = lane >> 4;   // lane half selects K sub-range

  v8f acc = {};

  // ---- product 1: A1 @ B1 ----
  {
    const __bf16* arow = A1 + (long)(m0 + r) * K1p;
    const __bf16* brow = B1T + (long)(n0 + r) * K1p;
    for (int kb = 0; kb < K1p; kb += 32) {
      union { v16bf v; v8bf q[2]; } a, b;
      // 16-bit A 16x32 layout: lane half h holds K = {h*8..h*8+7, 16+h*8..23+h*8}
      a.q[0] = *(const v8bf*)(arow + kb + h * 8);
      a.q[1] = *(const v8bf*)(arow + kb + 16 + h * 8);
      // B 32x16: lane half h holds K = {h*8..h*8+7, 16+h*8..23+h*8} of column n0+r
      b.q[0] = *(const v8bf*)(brow + kb + h * 8);
      b.q[1] = *(const v8bf*)(brow + kb + 16 + h * 8);
      acc = __builtin_amdgcn_wmma_f32_16x16x32_bf16(
          false, a.v, false, b.v, (short)0, acc, false, false);
    }
  }
  // ---- product 2: A2 @ B2 ----
  {
    const __bf16* arow = A2 + (long)(m0 + r) * K2p;
    const __bf16* brow = B2T + (long)(n0 + r) * K2p;
    for (int kb = 0; kb < K2p; kb += 32) {
      union { v16bf v; v8bf q[2]; } a, b;
      a.q[0] = *(const v8bf*)(arow + kb + h * 8);
      a.q[1] = *(const v8bf*)(arow + kb + 16 + h * 8);
      b.q[0] = *(const v8bf*)(brow + kb + h * 8);
      b.q[1] = *(const v8bf*)(brow + kb + 16 + h * 8);
      acc = __builtin_amdgcn_wmma_f32_16x16x32_bf16(
          false, a.v, false, b.v, (short)0, acc, false, false);
    }
  }

  // C/D layout: lane holds column n0+r; VGPR v holds row m0 + h*8 + v
  const float bn = bias[n0 + r];
  #pragma unroll
  for (int v = 0; v < 8; v++) {
    float val = acc[v] + bn;
    if (doRelu) val = fmaxf(val, 0.0f);
    long idx = (long)(m0 + h * 8 + v) * 256 + n0 + r;
    Cout[idx] = val;
    if (Cb) Cb[idx] = (__bf16)val;
  }
}

// ---------------- launch ----------------

static inline size_t align256(size_t x) { return (x + 255) & ~(size_t)255; }

extern "C" void kernel_launch(void* const* d_in, const int* in_sizes, int n_in,
                              void* d_out, int out_size, void* d_ws, size_t ws_size,
                              hipStream_t stream) {
  const float*     x    = (const float*)d_in[0];
  const long long* eidx = (const long long*)d_in[1];   // int64 [2, E]
  const float*     W1l  = (const float*)d_in[2];
  const float*     b1   = (const float*)d_in[3];
  const float*     W1r  = (const float*)d_in[4];
  const float*     W2l  = (const float*)d_in[5];
  const float*     b2   = (const float*)d_in[6];
  const float*     W2r  = (const float*)d_in[7];
  const long long* src  = eidx;
  const long long* dst  = eidx + NE;
  float* out = (float*)d_out;

  // workspace carve-up (256B aligned)
  char* ws = (char*)d_ws;
  size_t off = 0;
  float*  deg   = (float*)(ws + off); off = align256(off + (size_t)NN * 4);
  float*  agg   = (float*)(ws + off); off = align256(off + (size_t)NN * IN_CH * 4); // reused for layer2 (NN*256 fits)
  __bf16* meanb = (__bf16*)(ws + off); off = align256(off + (size_t)NN * IN_PAD * 2); // reused layer2
  __bf16* xb    = (__bf16*)(ws + off); off = align256(off + (size_t)NN * IN_PAD * 2);
  float*  hbuf  = (float*)(ws + off); off = align256(off + (size_t)NN * HID * 4);
  __bf16* hb    = (__bf16*)(ws + off); off = align256(off + (size_t)NN * HID * 2);
  __bf16* w1lT  = (__bf16*)(ws + off); off = align256(off + (size_t)HID * IN_PAD * 2);
  __bf16* w1rT  = (__bf16*)(ws + off); off = align256(off + (size_t)HID * IN_PAD * 2);
  __bf16* w2lT  = (__bf16*)(ws + off); off = align256(off + (size_t)HID * HID * 2);
  __bf16* w2rT  = (__bf16*)(ws + off); off = align256(off + (size_t)HID * HID * 2);
  (void)ws_size; (void)n_in; (void)in_sizes; (void)out_size;

  const int B = 256;
  auto cdiv = [](long a, long b) { return (int)((a + b - 1) / b); };

  // ---- layer 1 aggregation ----
  zero_f32<<<cdiv((long)NN * IN_CH, B), B, 0, stream>>>(agg, NN * IN_CH);
  zero_f32<<<cdiv(NN, B), B, 0, stream>>>(deg, NN);
  scatter_edges<<<cdiv((long)NE * 32, B), B, 0, stream>>>(x, src, dst, agg, deg,
                                                          NE, IN_CH, 1);
  make_invdeg<<<cdiv(NN, B), B, 0, stream>>>(deg, NN);

  // ---- bf16 conversions ----
  f32_to_bf16_pad<<<cdiv((long)NN * IN_PAD, B), B, 0, stream>>>(x, xb, IN_CH, IN_PAD, NN);
  mean_to_bf16<<<cdiv((long)NN * IN_PAD, B), B, 0, stream>>>(agg, deg, meanb, IN_CH, IN_PAD, NN);
  weight_T_bf16<<<cdiv((long)HID * IN_PAD, B), B, 0, stream>>>(W1l, w1lT, IN_CH, IN_PAD, HID);
  weight_T_bf16<<<cdiv((long)HID * IN_PAD, B), B, 0, stream>>>(W1r, w1rT, IN_CH, IN_PAD, HID);
  weight_T_bf16<<<cdiv((long)HID * HID, B), B, 0, stream>>>(W2l, w2lT, HID, HID, HID);
  weight_T_bf16<<<cdiv((long)HID * HID, B), B, 0, stream>>>(W2r, w2rT, HID, HID, HID);

  // ---- layer 1 GEMM: h = relu(mean@W1l + x@W1r + b1) ----
  dim3 g1(NN / 16, 2);
  sage_gemm<<<g1, 256, 0, stream>>>(meanb, w1lT, IN_PAD, xb, w1rT, IN_PAD,
                                    b1, hbuf, hb, 1);

  // ---- layer 2 aggregation (reuse agg/mean buffers) ----
  zero_f32<<<cdiv((long)NN * HID, B), B, 0, stream>>>(agg, NN * HID);
  scatter_edges<<<cdiv((long)NE * 32, B), B, 0, stream>>>(hbuf, src, dst, agg, deg,
                                                          NE, HID, 0);
  mean_to_bf16<<<cdiv((long)NN * HID, B), B, 0, stream>>>(agg, deg, meanb, HID, HID, NN);

  // ---- layer 2 GEMM: out = mean2@W2l + h@W2r + b2 ----
  dim3 g2(NN / 16, 2);
  sage_gemm<<<g2, 256, 0, stream>>>(meanb, w2lT, HID, hb, w2rT, HID,
                                    b2, out, (__bf16*)nullptr, 0);
}
